// SepNet_54211077210763
// MI455X (gfx1250) — compile-verified, matrix operand order checked by reading, not run on hardware
//
#include <hip/hip_runtime.h>
#include <hip/hip_bf16.h>

// ---------------------------------------------------------------------------
// SepNet for MI455X (gfx1250), wave32 + WMMA.
//   N=32768 nodes, F=64, H=4 heads, ES=EG=524288 edges, G=64 graphs, NC=10.
// GEMMs (x@W 64->256 and elu@mlpW 256->64) run on v_wmma_f32_16x16x32_f16
// with transposed+padded LDS panels so both A and B fragments load as
// ds_load_b128 (conflict-free).  Everything else is bandwidth-bound
// gather/scatter with f32 atomics.
// ---------------------------------------------------------------------------

#define N_C   32768
#define F_C   64
#define H_C   4
#define ES_C  524288
#define EG_C  524288
#define G_C   64
#define NC_C  10
#define E2_C  (EG_C + N_C)   // edges + self loops
#define BN_EPS 1e-5f

typedef _Float16 v16h __attribute__((ext_vector_type(16)));
typedef float    v8f  __attribute__((ext_vector_type(8)));

// ---------------------------------------------------------------------------
// WMMA GEMM: C[M,NN] = act(A[M,K] (+preBias)) @ B[K,NN] (+postBias)
// Block = 256 threads = 8 waves; each block computes an MB-row strip.
// A strip (MB x K) and B panel *transposed* (NN x K) staged in LDS as f16,
// rows padded by 8 halves so half-wave b128 reads are bank-conflict-free.
// Each wave owns one 16-row tile, preloads all A fragments into registers,
// then sweeps its column tiles loading only B fragments.
// HAS_PB is compile-time so no branch/cond-load pollutes the WMMA pipeline.
// Fragment layouts follow CDNA5 ISA §7.12.2 (16-bit A, 32-bit C/D).
// ---------------------------------------------------------------------------
template <int K, int NN, int MB, bool ELU, bool HAS_PB>
__global__ void __launch_bounds__(256)
gemm_wmma_kernel(const float* __restrict__ A,
                 const float* __restrict__ B,
                 const float* __restrict__ preBias,   // [K]  (ELU path)
                 const float* __restrict__ postBias,  // [NN] (HAS_PB path)
                 float* __restrict__ C)
{
    constexpr int KP = K + 8;            // padded row length (16B pad)
    __shared__ _Float16 As[MB][KP];
    __shared__ _Float16 Bs[NN][KP];      // B transposed: Bs[col][k]

    const int tid     = threadIdx.x;
    const int rowBase = blockIdx.x * MB;

    // Stage B panel transposed (f32 -> f16).  idx sweeps column-major so the
    // LDS writes per thread-group are contiguous; global reads stay L2-hot.
    for (int idx = tid; idx < K * NN; idx += 256) {
        const int col = idx / K, k = idx % K;
        Bs[col][k] = (_Float16)B[(long)k * NN + col];
    }
    // Stage A strip, optionally fused bias + ELU (GAT output -> MLP input)
    for (int idx = tid; idx < MB * K; idx += 256) {
        const int r = idx / K, k = idx % K;
        float v = A[(long)(rowBase + r) * K + k];
        if (ELU) {
            v += preBias[k];
            v = (v > 0.f) ? v : (__expf(v) - 1.f);
        }
        As[r][k] = (_Float16)v;
    }
    __syncthreads();

    const int wave = tid >> 5;
    const int lane = tid & 31;
    const int half = lane >> 4;          // 0: lanes 0-15, 1: lanes 16-31
    const int r    = lane & 15;

    constexpr int RT  = MB / 16;         // row tiles per block
    constexpr int WPR = 8 / RT;          // waves sharing one row tile
    constexpr int CPW = (NN / 16) / WPR; // column tiles per wave
    const int wr     = wave / WPR;                 // this wave's row tile
    const int wcBase = (wave % WPR) * CPW;         // first column tile

    // Preload all A fragments for this wave's 16-row strip.
    // 16-bit A layout: vgpr j/2 holds K pairs; lanes16-31 offset +8;
    // vgprs 4-7 cover K+16.  => k = k0 + (j/8)*16 + half*8 + (j%8)
    v16h afrag[K / 32];
#pragma unroll
    for (int ks = 0; ks < K / 32; ++ks) {
#pragma unroll
        for (int j = 0; j < 16; ++j) {
            const int kk = ks * 32 + ((j >> 3) << 4) + (half << 3) + (j & 7);
            afrag[ks][j] = As[wr * 16 + r][kk];
        }
    }

#pragma unroll
    for (int c = 0; c < CPW; ++c) {
        const int tc = (wcBase + c) * 16;
        v8f acc = {};
#pragma unroll
        for (int ks = 0; ks < K / 32; ++ks) {
            v16h b;
#pragma unroll
            for (int j = 0; j < 16; ++j) {
                const int kk = ks * 32 + ((j >> 3) << 4) + (half << 3) + (j & 7);
                b[j] = Bs[tc + r][kk];
            }
            acc = __builtin_amdgcn_wmma_f32_16x16x32_f16(
                      false, afrag[ks], false, b, (short)0, acc, false, false);
        }
        float pb = 0.f;
        if (HAS_PB) pb = postBias[tc + r];
#pragma unroll
        for (int v = 0; v < 8; ++v) {
            // 32-bit C/D layout: vgpr v -> row (v + 8*half), col = lane%16
            C[(long)(rowBase + wr * 16 + v + half * 8) * NN + (tc + r)] =
                acc[v] + pb;
        }
    }
}

// ---------------------------------------------------------------------------
// Scatter pass: out[dst] += attr * x[src]   (float4 payload, f32 atomics)
// One thread per (edge, 4-float chunk); F=64 -> 16 chunks per edge.
// ---------------------------------------------------------------------------
__global__ void scatter_pass_kernel(const int* __restrict__ src,
                                    const int* __restrict__ dst,
                                    const float* __restrict__ attr,
                                    const float* __restrict__ xin,
                                    float* __restrict__ out)
{
    const long t = (long)blockIdx.x * blockDim.x + threadIdx.x;
    const long total = (long)ES_C * 16;
    if (t >= total) return;
    const long e = t >> 4;
    const int  c = (int)(t & 15) * 4;
    const int  s = src[e], d = dst[e];
    const float a = attr[e];
    const float4 xv = *(const float4*)(xin + (long)s * F_C + c);
    float* o = out + (long)d * F_C + c;
    atomicAdd(o + 0, a * xv.x);
    atomicAdd(o + 1, a * xv.y);
    atomicAdd(o + 2, a * xv.z);
    atomicAdd(o + 3, a * xv.w);
}

__global__ void abs_kernel(float* __restrict__ p, long n)
{
    const long t = (long)blockIdx.x * blockDim.x + threadIdx.x;
    if (t < n) p[t] = fabsf(p[t]);
}

__global__ void fill_kernel(float* __restrict__ p, float v, long n)
{
    const long t = (long)blockIdx.x * blockDim.x + threadIdx.x;
    if (t < n) p[t] = v;
}

// ---------------------------------------------------------------------------
// GAT attention scores: a_s[n,h] = sum_f h[n,h,f]*asrc[h,f]; same for a_d.
// ---------------------------------------------------------------------------
__global__ void att_score_kernel(const float* __restrict__ h,
                                 const float* __restrict__ asrc,
                                 const float* __restrict__ adst,
                                 float* __restrict__ a_s,
                                 float* __restrict__ a_d)
{
    const long t = (long)blockIdx.x * blockDim.x + threadIdx.x;
    if (t >= (long)N_C * H_C) return;
    const int n  = (int)(t >> 2);
    const int hd = (int)(t & 3);
    const float* hp = h + (long)n * (H_C * F_C) + hd * F_C;
    const float* as = asrc + hd * F_C;
    const float* ad = adst + hd * F_C;
    float sa = 0.f, da = 0.f;
#pragma unroll 4
    for (int f = 0; f < F_C; ++f) {
        const float v = hp[f];
        sa += v * as[f];
        da += v * ad[f];
    }
    a_s[t] = sa;
    a_d[t] = da;
}

__device__ __forceinline__ void atomicMaxF(float* addr, float v)
{
    const int vi = __float_as_int(v);
    if (vi >= 0) atomicMax((int*)addr, vi);
    else         atomicMin((unsigned int*)addr, (unsigned int)vi);
}

__device__ __forceinline__ void edge_sd(const int* ei, int e, int& s, int& d)
{
    if (e < EG_C) { s = ei[e]; d = ei[EG_C + e]; }
    else          { s = d = e - EG_C; }         // self loop
}

// pass 1: segment max of leaky_relu scores
__global__ void edge_max_kernel(const int* __restrict__ ei,
                                const float* __restrict__ a_s,
                                const float* __restrict__ a_d,
                                float* __restrict__ m)
{
    const long t = (long)blockIdx.x * blockDim.x + threadIdx.x;
    if (t >= (long)E2_C * H_C) return;
    const int e = (int)(t >> 2), hd = (int)(t & 3);
    int s, d; edge_sd(ei, e, s, d);
    float sc = a_s[s * H_C + hd] + a_d[d * H_C + hd];
    sc = (sc > 0.f) ? sc : 0.2f * sc;
    atomicMaxF(&m[d * H_C + hd], sc);
}

// pass 2: ex = exp(score - m[dst]); denom[dst] += ex; stash ex per edge
__global__ void edge_exp_kernel(const int* __restrict__ ei,
                                const float* __restrict__ a_s,
                                const float* __restrict__ a_d,
                                const float* __restrict__ m,
                                float* __restrict__ exw,
                                float* __restrict__ denom)
{
    const long t = (long)blockIdx.x * blockDim.x + threadIdx.x;
    if (t >= (long)E2_C * H_C) return;
    const int e = (int)(t >> 2), hd = (int)(t & 3);
    int s, d; edge_sd(ei, e, s, d);
    float sc = a_s[s * H_C + hd] + a_d[d * H_C + hd];
    sc = (sc > 0.f) ? sc : 0.2f * sc;
    const float ex = __expf(sc - m[d * H_C + hd]);
    exw[t] = ex;
    atomicAdd(&denom[d * H_C + hd], ex);
}

// pass 3: g[dst] += alpha * h[src]   (float4 payload; one thread per (e, chunk))
__global__ void edge_agg_kernel(const int* __restrict__ ei,
                                const float* __restrict__ exw,
                                const float* __restrict__ denom,
                                const float* __restrict__ h,
                                float* __restrict__ g)
{
    const long t = (long)blockIdx.x * blockDim.x + threadIdx.x;
    const long total = (long)E2_C * 64;   // 256 floats = 64 float4 per edge
    if (t >= total) return;
    const int e  = (int)(t >> 6);
    const int c4 = (int)(t & 63);
    const int hd = c4 >> 4;
    const int col = c4 * 4;
    int s, d; edge_sd(ei, e, s, d);
    const float alpha = exw[(long)e * H_C + hd] / denom[d * H_C + hd];
    const float4 hv = *(const float4*)(h + (long)s * 256 + col);
    float* o = g + (long)d * 256 + col;
    atomicAdd(o + 0, alpha * hv.x);
    atomicAdd(o + 1, alpha * hv.y);
    atomicAdd(o + 2, alpha * hv.z);
    atomicAdd(o + 3, alpha * hv.w);
}

// ---------------------------------------------------------------------------
// BatchNorm over N rows, 64 columns: stats then fused apply + residual add.
// ---------------------------------------------------------------------------
__global__ void bn_stats_kernel(const float* __restrict__ z,
                                float* __restrict__ stats)   // [0:64)=mu [64:128)=var
{
    __shared__ float s1[256], s2[256];
    const int col = blockIdx.x;
    const int tid = threadIdx.x;
    float a = 0.f, b = 0.f;
    for (int r = tid; r < N_C; r += 256) {
        const float v = z[(long)r * F_C + col];
        a += v; b += v * v;
    }
    s1[tid] = a; s2[tid] = b;
    __syncthreads();
    for (int s = 128; s > 0; s >>= 1) {
        if (tid < s) { s1[tid] += s1[tid + s]; s2[tid] += s2[tid + s]; }
        __syncthreads();
    }
    if (tid == 0) {
        const float mu = s1[0] / (float)N_C;
        stats[col]      = mu;
        stats[64 + col] = s2[0] / (float)N_C - mu * mu;
    }
}

__global__ void bn_apply_add_kernel(const float* __restrict__ z,
                                    const float* __restrict__ stats,
                                    const float* __restrict__ gw,
                                    const float* __restrict__ bw,
                                    float* __restrict__ acc)
{
    const long t = (long)blockIdx.x * blockDim.x + threadIdx.x;
    if (t >= (long)N_C * F_C) return;
    const int col = (int)(t & (F_C - 1));
    const float mu = stats[col], var = stats[64 + col];
    acc[t] += (z[t] - mu) * rsqrtf(var + BN_EPS) * gw[col] + bw[col];
}

// ---------------------------------------------------------------------------
// global_add_pool over sorted batch assignment.
// ---------------------------------------------------------------------------
__global__ void pool_kernel(const float* __restrict__ acc,
                            const int* __restrict__ batch,
                            float* __restrict__ pooled)
{
    const long t = (long)blockIdx.x * blockDim.x + threadIdx.x;
    if (t >= (long)N_C * 16) return;
    const int n = (int)(t >> 4);
    const int c = (int)(t & 15) * 4;
    const int g = batch[n];
    const float4 v = *(const float4*)(acc + (long)n * F_C + c);
    float* o = pooled + (long)g * F_C + c;
    atomicAdd(o + 0, v.x);
    atomicAdd(o + 1, v.y);
    atomicAdd(o + 2, v.z);
    atomicAdd(o + 3, v.w);
}

// ---------------------------------------------------------------------------
// Head (G=64 rows, tiny): single-block GEMM + single-block BN(+ReLU).
// ---------------------------------------------------------------------------
__global__ void head_gemm_kernel(const float* __restrict__ X,
                                 const float* __restrict__ W,
                                 const float* __restrict__ b,
                                 float* __restrict__ Y,
                                 int Kdim, int Nout)
{
    for (int idx = threadIdx.x; idx < G_C * Nout; idx += blockDim.x) {
        const int row = idx / Nout, col = idx % Nout;
        float acc = b[col];
        for (int k = 0; k < Kdim; ++k)
            acc += X[row * Kdim + k] * W[k * Nout + col];
        Y[idx] = acc;
    }
}

__global__ void head_bn_kernel(const float* __restrict__ Y,
                               const float* __restrict__ g,
                               const float* __restrict__ beta,
                               float* __restrict__ out,
                               int Nout, int relu)
{
    for (int col = threadIdx.x; col < Nout; col += blockDim.x) {
        float a = 0.f, b = 0.f;
        for (int r = 0; r < G_C; ++r) {
            const float v = Y[r * Nout + col];
            a += v; b += v * v;
        }
        const float mu  = a / (float)G_C;
        const float var = b / (float)G_C - mu * mu;
        const float inv = rsqrtf(var + BN_EPS) * g[col];
        for (int r = 0; r < G_C; ++r) {
            float v = (Y[r * Nout + col] - mu) * inv + beta[col];
            if (relu) v = fmaxf(v, 0.f);
            out[r * Nout + col] = v;
        }
    }
}

// ---------------------------------------------------------------------------
// Launch
// ---------------------------------------------------------------------------
static inline int nblk(long total, int bs) { return (int)((total + bs - 1) / bs); }

extern "C" void kernel_launch(void* const* d_in, const int* in_sizes, int n_in,
                              void* d_out, int out_size, void* d_ws, size_t ws_size,
                              hipStream_t stream)
{
    (void)in_sizes; (void)n_in; (void)out_size; (void)ws_size;

    const float* x       = (const float*)d_in[0];
    const int*   ei      = (const int*)  d_in[1];
    const int*   batch   = (const int*)  d_in[2];
    const int*   sidx    = (const int*)  d_in[3];   // [4][2][ES]
    const float* sattr   = (const float*)d_in[4];   // [4][ES]
    const float* gat_W   = (const float*)d_in[5];   // [4][64][256]
    const float* gat_as  = (const float*)d_in[6];   // [4][4][64]
    const float* gat_ad  = (const float*)d_in[7];
    const float* gat_b   = (const float*)d_in[8];   // [4][256]
    const float* mlp_W   = (const float*)d_in[9];   // [4][256][64]
    const float* mlp_b   = (const float*)d_in[10];  // [4][64]
    const float* mlp_g   = (const float*)d_in[11];
    const float* mlp_bt  = (const float*)d_in[12];
    const float* fc1_W   = (const float*)d_in[13];
    const float* fc1_b   = (const float*)d_in[14];
    const float* fc1_g   = (const float*)d_in[15];
    const float* fc1_bt  = (const float*)d_in[16];
    const float* fc2_W   = (const float*)d_in[17];
    const float* fc2_b   = (const float*)d_in[18];
    const float* fc2_g   = (const float*)d_in[19];
    const float* fc2_bt  = (const float*)d_in[20];
    const float* fc3_W   = (const float*)d_in[21];
    const float* fc3_b   = (const float*)d_in[22];
    const float* fc3_g   = (const float*)d_in[23];
    const float* fc3_bt  = (const float*)d_in[24];
    float* out = (float*)d_out;

    // ---- workspace carve (floats) ----
    const size_t NF = (size_t)N_C * F_C;           // 2,097,152
    float* w = (float*)d_ws;
    float* xb[4]; for (int i = 0; i < 4; ++i) { xb[i] = w; w += NF; }
    float* tmp    = w; w += NF;
    float* hbuf   = w; w += (size_t)N_C * 256;
    float* gbuf   = w; w += (size_t)N_C * 256;
    float* zbuf   = w; w += NF;
    float* acc    = w; w += NF;
    float* a_s    = w; w += (size_t)N_C * H_C;
    float* a_d    = w; w += (size_t)N_C * H_C;
    float* mbuf   = w; w += (size_t)N_C * H_C;
    float* denom  = w; w += (size_t)N_C * H_C;
    float* exw    = w; w += (size_t)E2_C * H_C;
    float* stats  = w; w += 256;
    float* pooled = w; w += (size_t)G_C * F_C;
    float* hb1    = w; w += (size_t)G_C * 256;
    float* hb2    = w; w += (size_t)G_C * 128;

    const int BS = 256;

    // ================= Stage 0: scatter passes =================
    // x0..x3 and tmp are contiguous -> one memset
    hipMemsetAsync(xb[0], 0, 5 * NF * sizeof(float), stream);

    const long scat_total = (long)ES_C * 16;
    // x0 = scatter(idx0, attr0, x)
    scatter_pass_kernel<<<nblk(scat_total, BS), BS, 0, stream>>>(
        sidx + 0, sidx + ES_C, sattr + 0, x, xb[0]);
    for (int i = 1; i < 4; ++i) {
        hipMemsetAsync(tmp, 0, NF * sizeof(float), stream);
        scatter_pass_kernel<<<nblk(scat_total, BS), BS, 0, stream>>>(
            sidx + (size_t)i * 2 * ES_C, sidx + (size_t)i * 2 * ES_C + ES_C,
            sattr + (size_t)i * ES_C, x, tmp);
        abs_kernel<<<nblk((long)NF, BS), BS, 0, stream>>>(tmp, (long)NF);
        scatter_pass_kernel<<<nblk(scat_total, BS), BS, 0, stream>>>(
            sidx + 0, sidx + ES_C, sattr + 0, tmp, xb[i]);
    }

    // acc = x (residual accumulator: x + sum_i BN(z_i))
    hipMemcpyAsync(acc, x, NF * sizeof(float), hipMemcpyDeviceToDevice, stream);

    // ================= Stage 1: 4 independent GAT branches =================
    for (int i = 0; i < 4; ++i) {
        // h = x_i @ gat_W[i]    (WMMA, 64 -> 256, 64-row strips)
        gemm_wmma_kernel<64, 256, 64, false, false><<<N_C / 64, BS, 0, stream>>>(
            xb[i], gat_W + (size_t)i * 64 * 256, nullptr, nullptr, hbuf);

        // attention logits per node/head
        att_score_kernel<<<nblk((long)N_C * H_C, BS), BS, 0, stream>>>(
            hbuf, gat_as + (size_t)i * H_C * F_C, gat_ad + (size_t)i * H_C * F_C,
            a_s, a_d);

        // segment softmax over incoming edges (+ self loops)
        fill_kernel<<<nblk((long)N_C * H_C, BS), BS, 0, stream>>>(
            mbuf, -1e30f, (long)N_C * H_C);
        edge_max_kernel<<<nblk((long)E2_C * H_C, BS), BS, 0, stream>>>(
            ei, a_s, a_d, mbuf);
        hipMemsetAsync(denom, 0, (size_t)N_C * H_C * sizeof(float), stream);
        edge_exp_kernel<<<nblk((long)E2_C * H_C, BS), BS, 0, stream>>>(
            ei, a_s, a_d, mbuf, exw, denom);

        // aggregate: g[dst] += alpha * h[src]
        hipMemsetAsync(gbuf, 0, (size_t)N_C * 256 * sizeof(float), stream);
        edge_agg_kernel<<<nblk((long)E2_C * 64, BS), BS, 0, stream>>>(
            ei, exw, denom, hbuf, gbuf);

        // z = elu(g + gat_b[i]) @ mlp_W[i] + mlp_b[i]  (WMMA, 256 -> 64, 32-row)
        gemm_wmma_kernel<256, 64, 32, true, true><<<N_C / 32, BS, 0, stream>>>(
            gbuf, mlp_W + (size_t)i * 256 * 64,
            gat_b + (size_t)i * 256, mlp_b + (size_t)i * 64, zbuf);

        // BatchNorm + residual accumulate
        bn_stats_kernel<<<64, BS, 0, stream>>>(zbuf, stats);
        bn_apply_add_kernel<<<nblk((long)NF, BS), BS, 0, stream>>>(
            zbuf, stats, mlp_g + (size_t)i * 64, mlp_bt + (size_t)i * 64, acc);
    }

    // ================= Stage 2: pooling + head =================
    hipMemsetAsync(pooled, 0, (size_t)G_C * F_C * sizeof(float), stream);
    pool_kernel<<<nblk((long)N_C * 16, BS), BS, 0, stream>>>(acc, batch, pooled);

    head_gemm_kernel<<<1, BS, 0, stream>>>(pooled, fc1_W, fc1_b, hb1, 64, 256);
    head_bn_kernel  <<<1, BS, 0, stream>>>(hb1, fc1_g, fc1_bt, hb1, 256, 1);
    head_gemm_kernel<<<1, BS, 0, stream>>>(hb1, fc2_W, fc2_b, hb2, 256, 128);
    head_bn_kernel  <<<1, BS, 0, stream>>>(hb2, fc2_g, fc2_bt, hb2, 128, 1);
    head_gemm_kernel<<<1, BS, 0, stream>>>(hb2, fc3_W, fc3_b, tmp, 128, NC_C);
    head_bn_kernel  <<<1, BS, 0, stream>>>(tmp, fc3_g, fc3_bt, out, NC_C, 0);
}
